// CollaborativeLearningModule_69492570849959
// MI455X (gfx1250) — matrix operand order repaired and checked by reading, not compiled, not run on hardware
//
#include <hip/hip_runtime.h>
#include <hip/hip_bf16.h>

// ======================================================================
// CollaborativeLearningModule forward for MI455X (gfx1250, wave32, WMMA).
//
// - All conv / deform-einsum / attention GEMMs: v_wmma_f32_16x16x32_f16
//   (f16 multiply, f32 accumulate), fragments per ISA 7.12.2 layouts.
// - Convs: 4-wave blocks stage the (Cin x ksz x Wt) input tile in LDS as
//   f16 once, shared by 4 cout tiles (4x reuse, coalesced global loads,
//   ds_load fragment gathers).
// - Deform einsum (1x1, K=576, no halo): tile DMA'd into LDS by the
//   Tensor Data Mover (tensor_load_to_lds + s_wait_tensorcnt), 6-arg
//   builtin form (clang-23 / therock-10.0), with a cooperative-load
//   fallback if the builtin is unavailable.
// - Roofline: ~40 GFLOP total, ~60 MB live activations -> fully L2
//   resident (192 MB), so the win is WMMA throughput + LDS reuse, not HBM.
//
// Workspace assumption: ws_size >= ~201 MB (1362 planes of 192*192 f32).
// ======================================================================

#define HW   192
#define HW2  (HW * HW)

typedef __attribute__((ext_vector_type(16))) _Float16 v16h;
typedef __attribute__((ext_vector_type(8)))  float    v8f;

#if __has_builtin(__builtin_amdgcn_tensor_load_to_lds)
#define HAVE_TDM 1
typedef __attribute__((ext_vector_type(4))) unsigned int tdm_u32x4;
typedef __attribute__((ext_vector_type(8))) int          tdm_i32x8;
typedef __attribute__((ext_vector_type(4))) int          tdm_i32x4;
#endif

// 16-bit A fragment (16x32, MxK): lanes 0-15 row m=lane, halves 0..7 ->
// K 0..7, halves 8..15 -> K 16..23; lanes 16-31 same rows, K offset +8.
__device__ __forceinline__ int a_kmap(int lane, int i) {
    return i + ((i >> 3) << 3) + ((lane >> 4) << 3);
}
// 16-bit B fragment (32x16, KxN): lanes 0-15 col n=lane, halves 0..7 ->
// K 0..7, halves 8..15 -> K 16..23; lanes 16-31 col n=lane-16, K offset +8.
__device__ __forceinline__ int b_kmap(int lane, int i) {
    return i + ((i >> 3) << 3) + ((lane >> 4) << 3);
}

// ----------------------------------------------------------------------
// Generic implicit-GEMM convolution.
// Block = 128 threads (4 waves). Block owns one 16-pixel strip of one
// output row and 64 output channels (one 16-ch tile per wave).  The
// input tile (Cin x ksz rows x Wt cols, Wt = 16+(ksz-1)*dil) is staged
// cooperatively into LDS as f16, then each wave runs the tap/K-chunk
// WMMA loop out of LDS.  Input may be a virtual concat in0(C0)||in1(C1),
// C0 always a multiple of 32 so K-chunks never straddle the seam.
// Epilogue fuses +bias, BN affine, ReLU/sigmoid, +residual.
// act: 0=none, 1=relu, 2=sigmoid
// ----------------------------------------------------------------------
__global__ __launch_bounds__(128)
void conv_wmma_kernel(const float* __restrict__ in0, int C0,
                      const float* __restrict__ in1, int C1,
                      const float* __restrict__ wgt,
                      const float* __restrict__ bias,
                      const float* __restrict__ bng, const float* __restrict__ bnb,
                      const float* __restrict__ bnm, const float* __restrict__ bnv,
                      const float* __restrict__ res,
                      float* __restrict__ out,
                      int Cout, int ksz, int pad, int dil, int act)
{
    __shared__ _Float16 tile[6912];      // max: Cin=128, ksz=3, Wt=18

    const int tid  = threadIdx.x;
    const int lane = tid & 31;
    const int wv   = tid >> 5;
    const int blk  = blockIdx.x;
    const int xt   = blk % (HW / 16);
    const int py   = (blk / (HW / 16)) % HW;
    const int ctg  = blk / ((HW / 16) * HW);
    const int x0   = xt * 16;
    const int Cin  = C0 + C1;
    const int Wt   = 16 + (ksz - 1) * dil;
    const int m    = lane & 15;
    const int n    = (ctg * 4 + wv) * 16 + m;   // output channel of this lane

    // ---- stage input tile: tile[(ch*ksz + ky)*Wt + col] ----
    const int nelem = Cin * ksz * Wt;
    for (int e = tid; e < nelem; e += 128) {
        const int col = e % Wt;
        const int r   = e / Wt;
        const int ky  = r % ksz;
        const int ch  = r / ksz;
        const int y   = py + ky * dil - pad;
        const int x   = x0 - pad + col;
        float v = 0.f;
        if (y >= 0 && y < HW && x >= 0 && x < HW) {
            const int pix = y * HW + x;
            if (ch < C0) v = in0[ch * HW2 + pix];
            else         v = in1[(ch - C0) * HW2 + pix];
        }
        tile[e] = (_Float16)v;
    }
    __syncthreads();

    // ---- WMMA main loop (EXEC all ones) ----
    v8f acc = {};
    for (int tap = 0; tap < ksz * ksz; ++tap) {
        const int ky = tap / ksz, kx = tap % ksz;
        const int colbase = m + kx * dil;
        __builtin_prefetch(&wgt[n * Cin * (ksz * ksz) + tap], 0, 3);
        for (int kc = 0; kc < Cin; kc += 32) {
            v16h af, bf;
#pragma unroll
            for (int i = 0; i < 16; ++i) {
                const int ch = kc + a_kmap(lane, i);
                af[i] = tile[(ch * ksz + ky) * Wt + colbase];
            }
#pragma unroll
            for (int i = 0; i < 16; ++i) {
                const int ci = kc + b_kmap(lane, i);
                float wv2 = 0.f;
                if (n < Cout) wv2 = wgt[(n * Cin + ci) * (ksz * ksz) + tap];
                bf[i] = (_Float16)wv2;
            }
            acc = __builtin_amdgcn_wmma_f32_16x16x32_f16(false, af, false, bf,
                                                         (short)0, acc, false, false);
        }
    }

    // ---- epilogue ----
    if (n < Cout) {
        float sc = 1.f, sh = 0.f;
        if (bng) {
            const float inv = bng[n] * rsqrtf(bnv[n] + 1e-5f);
            sc = inv;
            sh = bnb[n] - bnm[n] * inv;
        }
        const float bs = bias ? bias[n] : 0.f;
#pragma unroll
        for (int j = 0; j < 8; ++j) {
            const int mm = j + ((lane >> 4) << 3);        // D: row = j / j+8
            float v = (acc[j] + bs) * sc + sh;
            if (act == 1)      v = fmaxf(v, 0.f);
            else if (act == 2) v = 1.f / (1.f + __expf(-v));
            const int oidx = n * HW2 + py * HW + x0 + mm;
            if (res) v += res[oidx];
            out[oidx] = v;
        }
    }
}

// ----------------------------------------------------------------------
// Deform einsum as 1x1 GEMM, K = 576 (= 64 ch * 9 taps), Cout = 64.
// The (576 x 16) f32 input tile is a perfectly regular 2D tile:
// dim0 = 16 contiguous floats (pixels), dim1 = 576 rows, stride HW2.
// Wave 0 issues one Tensor Data Mover descriptor to pull it into LDS
// (tracked by TENSORcnt), then 4 waves run 18 K-chunk WMMAs each.
// ----------------------------------------------------------------------
__global__ __launch_bounds__(128)
void deform_gemm_kernel(const float* __restrict__ s,     // 576 planes
                        const float* __restrict__ wgt,   // [64][576] flat OIHW
                        const float* __restrict__ bias,
                        float* __restrict__ out)
{
    __shared__ float tileF[576 * 16];                    // 36 KB

    const int tid  = threadIdx.x;
    const int lane = tid & 31;
    const int wv   = tid >> 5;
    const int blk  = blockIdx.x;
    const int x0   = (blk % (HW / 16)) * 16;
    const int py   = blk / (HW / 16);
    const int m    = lane & 15;
    const int n    = wv * 16 + m;                        // out channel

#if defined(HAVE_TDM)
    if (wv == 0) {
        // D# group 0: count=1 | lds_addr | global_addr | type=2  (ISA 8.3)
        const unsigned long long ga =
            (unsigned long long)(uintptr_t)s +
            (unsigned long long)(py * HW + x0) * 4ull;
        tdm_u32x4 g0;
        g0.x = 1u;                                       // count=1 (valid user D#)
        g0.y = (unsigned int)(uintptr_t)(void*)tileF;    // lds_addr (LDS offset)
        g0.z = (unsigned int)ga;                         // global_addr[31:0]
        g0.w = ((unsigned int)(ga >> 32) & 0x01ffffffu)  // global_addr[56:32]
               | (2u << 30);                             // type=2 ("image")
        // D# group 1: data_size=4B, tensor_dim0=16, tensor_dim1=576,
        // tile_dim0=16, tile_dim1=576, dim0_stride=HW2     (ISA 8.4)
        tdm_i32x8 g1;
        g1[0] = 0x00020000;           // wg_mask=0 | data_size=2 (4 bytes)
        g1[1] = (int)(16u << 16);     // atomic_barrier_addr=0 | tensor_dim0[15:0]=16
        g1[2] = (int)(576u << 16);    // tensor_dim0[31:16]=0 | tensor_dim1[15:0]=576
        g1[3] = (int)(16u << 16);     // tensor_dim1[31:16]=0 | tile_dim0=16
        g1[4] = 576;                  // tile_dim1=576 | tile_dim2=0
        g1[5] = HW2;                  // tensor_dim0_stride[31:0] = 36864
        g1[6] = 0;                    // stride[47:32]=0 | dim1_stride lo=0
        g1[7] = 0;
        tdm_i32x4 gz4 = {0, 0, 0, 0};                    // groups 2/3 unused (2D)
        tdm_i32x8 gz8 = {0, 0, 0, 0, 0, 0, 0, 0};        // extra group (6-arg form)
        __builtin_amdgcn_tensor_load_to_lds(g0, g1, gz4, gz4, gz8, 0);
        __builtin_amdgcn_s_wait_tensorcnt(0);
    }
#else
    for (int e = tid; e < 576 * 16; e += 128)
        tileF[e] = s[(e >> 4) * HW2 + py * HW + x0 + (e & 15)];
#endif
    __syncthreads();

    v8f acc = {};
    for (int kc = 0; kc < 576; kc += 32) {
        v16h af, bf;
#pragma unroll
        for (int i = 0; i < 16; ++i)
            af[i] = (_Float16)tileF[(kc + a_kmap(lane, i)) * 16 + m];
#pragma unroll
        for (int i = 0; i < 16; ++i)
            bf[i] = (_Float16)wgt[n * 576 + kc + b_kmap(lane, i)];
        acc = __builtin_amdgcn_wmma_f32_16x16x32_f16(false, af, false, bf,
                                                     (short)0, acc, false, false);
    }

    const float bs = bias[n];
#pragma unroll
    for (int j = 0; j < 8; ++j) {
        const int mm = j + ((lane >> 4) << 3);
        out[n * HW2 + py * HW + x0 + mm] = acc[j] + bs;
    }
}

// ----------------------------------------------------------------------
// Deformable-conv bilinear sampling (torchvision semantics: k=3, pad=1,
// 1 deformable group).  Writes s[(c*9+k)*HW2 + p] so the einsum above is
// exactly a 1x1 GEMM with flat OIHW weights (o*576 + c*9 + k).
// ----------------------------------------------------------------------
__global__ void deform_sample_kernel(const float* __restrict__ x,
                                     const float* __restrict__ off,
                                     float* __restrict__ s)
{
    const int idx = blockIdx.x * blockDim.x + threadIdx.x;
    if (idx >= 9 * HW2) return;
    const int p = idx % HW2;
    const int k = idx / HW2;
    const int h = p / HW, w = p % HW;
    const float py = (float)(h - 1 + (k / 3)) + off[(k * 2 + 0) * HW2 + p];
    const float px = (float)(w - 1 + (k % 3)) + off[(k * 2 + 1) * HW2 + p];
    const float y0f = floorf(py), x0f = floorf(px);
    const float dy = py - y0f, dx = px - x0f;
    const int y0 = (int)y0f, x0 = (int)x0f;
    const float wt[4] = {(1.f - dy) * (1.f - dx), (1.f - dy) * dx,
                         dy * (1.f - dx),         dy * dx};
    const int ys[4] = {y0, y0, y0 + 1, y0 + 1};
    const int xs[4] = {x0, x0 + 1, x0, x0 + 1};
    for (int c = 0; c < 64; ++c) {
        const float* xc = x + c * HW2;
        float a = 0.f;
#pragma unroll
        for (int q = 0; q < 4; ++q) {
            const int yy = ys[q], xx = xs[q];
            if (yy >= 0 && yy < HW && xx >= 0 && xx < HW)
                a += wt[q] * xc[yy * HW + xx];
        }
        s[(c * 9 + k) * HW2 + p] = a;
    }
}

// ----------------------------------------------------------------------
// Window attention, one block (4 waves) per 64x64 window, replicating the
// reference's part/rev index permutations in the LDS gather / global
// scatter.  S = softmax(Q K^T / 8), O = S V; both GEMMs on WMMA f16.
// 1/rowsum is folded into the output scale so P is stored unnormalized.
// ----------------------------------------------------------------------
__global__ __launch_bounds__(128)
void win_attn_kernel(const float* __restrict__ qbuf,   // q = channels 0..63
                     const float* __restrict__ kvbuf,  // k = ch 64..127, v = ch 128..191
                     float* __restrict__ outbuf)       // 64 channels
{
    __shared__ _Float16 Qs[64][64];
    __shared__ _Float16 Ks[64][64];
    __shared__ _Float16 Vs[64][64];
    __shared__ float    Ss[4][16][64];
    __shared__ _Float16 Ps[4][16][64];
    __shared__ float    Rs[4][16];

    const int tid  = threadIdx.x;
    const int lane = tid & 31;
    const int wv   = tid >> 5;
    const int nw   = blockIdx.x;

    // part(): window n, token t, feat f  ->  (c,h,w) of the NCHW map
    for (int e = tid; e < 4096; e += 128) {
        const int t = e >> 6, f = e & 63;
        const int L = nw * 4096 + e;
        const int h = L % HW;
        const int M = L / HW;
        const int c = (((M >> 6) & 7) << 3) + (M & 7);
        const int w = ((M >> 9) << 3) + ((M >> 3) & 7);
        const int pix = c * HW2 + h * HW + w;
        Qs[t][f] = (_Float16)qbuf[pix];
        Ks[t][f] = (_Float16)kvbuf[64 * HW2 + pix];
        Vs[t][f] = (_Float16)kvbuf[128 * HW2 + pix];
    }
    __syncthreads();

    const int m = lane & 15;

    // S = Q K^T * 0.125 ; this wave owns token rows [wv*16, wv*16+16)
#pragma unroll
    for (int j = 0; j < 4; ++j) {
        v8f acc = {};
#pragma unroll
        for (int kc = 0; kc < 64; kc += 32) {
            v16h af, bf;
#pragma unroll
            for (int i = 0; i < 16; ++i)
                af[i] = Qs[wv * 16 + m][kc + a_kmap(lane, i)];
#pragma unroll
            for (int i = 0; i < 16; ++i)
                bf[i] = Ks[j * 16 + m][kc + b_kmap(lane, i)];   // B = K^T
            acc = __builtin_amdgcn_wmma_f32_16x16x32_f16(false, af, false, bf,
                                                         (short)0, acc, false, false);
        }
#pragma unroll
        for (int v = 0; v < 8; ++v)
            Ss[wv][v + ((lane >> 4) << 3)][j * 16 + m] = acc[v] * 0.125f;
    }
    __syncthreads();

    // softmax rows (lanes 0-15 each own one row of this wave's strip)
    if (lane < 16) {
        float mx = -1e30f;
        for (int i = 0; i < 64; ++i) mx = fmaxf(mx, Ss[wv][lane][i]);
        float sum = 0.f;
        for (int i = 0; i < 64; ++i) {
            const float ev = __expf(Ss[wv][lane][i] - mx);
            sum += ev;
            Ps[wv][lane][i] = (_Float16)ev;
        }
        Rs[wv][lane] = 1.f / sum;
    }
    __syncthreads();

    // O = (P / rowsum) V, scattered through rev()
#pragma unroll
    for (int j = 0; j < 4; ++j) {
        v8f acc = {};
#pragma unroll
        for (int kc = 0; kc < 64; kc += 32) {
            v16h af, bf;
#pragma unroll
            for (int i = 0; i < 16; ++i)
                af[i] = Ps[wv][m][kc + a_kmap(lane, i)];
#pragma unroll
            for (int i = 0; i < 16; ++i)
                bf[i] = Vs[kc + b_kmap(lane, i)][j * 16 + m];
            acc = __builtin_amdgcn_wmma_f32_16x16x32_f16(false, af, false, bf,
                                                         (short)0, acc, false, false);
        }
#pragma unroll
        for (int v = 0; v < 8; ++v) {
            const int r  = v + ((lane >> 4) << 3);
            const int t  = wv * 16 + r;
            const int f  = j * 16 + m;
            const float val = acc[v] * Rs[wv][r];
            const int M2 = nw * 64 + t;                 // rev(): c = f
            const int hh = (M2 / 1536) * 8 + ((M2 >> 3) & 7);
            const int ww = ((M2 >> 6) % 24) * 8 + (M2 & 7);
            outbuf[f * HW2 + hh * HW + ww] = val;
        }
    }
}

// ----------------------------------------------------------------------
// Host-side driver
// ----------------------------------------------------------------------
static inline void launch_conv(hipStream_t st,
                               const float* in0, int C0, const float* in1, int C1,
                               const float* w, const float* bias,
                               const float* g, const float* b, const float* mm, const float* vv,
                               const float* res, float* out,
                               int Cout, int ks, int pad, int dil, int act)
{
    const int ctgroups = (Cout + 63) >> 6;     // 64 out-channels per block
    const int blocks = (HW / 16) * HW * ctgroups;
    conv_wmma_kernel<<<blocks, 128, 0, st>>>(in0, C0, in1, C1, w, bias,
                                             g, b, mm, vv, res, out,
                                             Cout, ks, pad, dil, act);
}

extern "C" void kernel_launch(void* const* d_in, const int* in_sizes, int n_in,
                              void* d_out, int out_size, void* d_ws, size_t ws_size,
                              hipStream_t stream)
{
    (void)in_sizes; (void)n_in; (void)out_size; (void)ws_size;
    const float* Up = (const float*)d_in[0];
    const float* Uc = (const float*)d_in[1];
#define P(i) ((const float*)d_in[(i)])

    // ---- workspace layout (floats) ----
    float* W = (float*)d_ws;
    const size_t CP = (size_t)HW2;
    float* t1   = W; W += 32  * CP;
    float* t2   = W; W += 32  * CP;
    float* offb = W; W += 18  * CP;
    float* defb = W; W += 64  * CP;
    float* qA   = W; W += 192 * CP;
    float* qB   = W; W += 192 * CP;
    float* oA   = W; W += 64  * CP;
    float* oB   = W; W += 64  * CP;
    float* u1   = W; W += 64  * CP;
    float* u2   = W; W += 64  * CP;
    float* sdef = W; W += 576 * CP;

    // ---- output layout: (UP, UC, FP, FC, classfi, pansh) flat ----
    float* out = (float*)d_out;
    const size_t N = 64 * CP;
    float* UPo = out + 0 * N;
    float* UCo = out + 1 * N;
    float* FP  = out + 2 * N;
    float* FC  = out + 3 * N;
    float* CLS = out + 4 * N;
    float* PSH = out + 5 * N;

    // ---- fc_to_fp(U_classify) -> FP ----
    launch_conv(stream, Uc, 64, 0, 0, P(2), P(3), 0,0,0,0, 0, t1, 32, 3,1,1, 1);
    launch_conv(stream, t1, 32, 0, 0, P(4), P(5), P(16),P(17),P(18),P(19), 0, t2, 32, 3,1,1, 1);
    launch_conv(stream, t2, 32, 0, 0, P(6), P(7), P(20),P(21),P(22),P(23), 0, t1, 32, 3,2,2, 1);
    launch_conv(stream, t1, 32, 0, 0, P(8), P(9), 0,0,0,0, 0, FP, 64, 1,0,1, 2);

    // ---- fp_to_fc(U_pansharp) -> FC ----
    launch_conv(stream, Up, 64, 0, 0, P(10), P(11), 0,0,0,0, 0, t2, 32, 5,2,1, 1);
    launch_conv(stream, t2, 32, 0, 0, P(12), P(13), P(24),P(25),P(26),P(27), 0, t1, 32, 3,1,1, 1);
    launch_conv(stream, t1, 32, 0, 0, P(14), P(15), 0,0,0,0, 0, FC, 64, 3,1,1, 1);

    const int dsBlocks = (9 * HW2 + 255) / 256;
    const int dgBlocks = (HW / 16) * HW;

    // ---- deform branch 1: off1 -> FC_def -> classfi ----
    launch_conv(stream, FP, 64, FC, 64, P(28), P(29), 0,0,0,0, 0, offb, 18, 3,1,1, 0);
    deform_sample_kernel<<<dsBlocks, 256, 0, stream>>>(FC, offb, sdef);
    deform_gemm_kernel<<<dgBlocks, 128, 0, stream>>>(sdef, P(32), P(33), defb);
    launch_conv(stream, FP, 64, defb, 64, P(36), P(37), 0,0,0,0, 0, CLS, 64, 3,1,1, 0);

    // ---- deform branch 2: off2 -> FP_def -> pansh ----
    launch_conv(stream, FP, 64, FC, 64, P(30), P(31), 0,0,0,0, 0, offb, 18, 3,1,1, 0);
    deform_sample_kernel<<<dsBlocks, 256, 0, stream>>>(FP, offb, sdef);
    deform_gemm_kernel<<<dgBlocks, 128, 0, stream>>>(sdef, P(34), P(35), defb);
    launch_conv(stream, FC, 64, defb, 64, P(38), P(39), 0,0,0,0, 0, PSH, 64, 3,1,1, 0);

    // ---- cross_attn ca1: xA = classfi, xB = U_classify -> UC ----
    launch_conv(stream, CLS, 64, 0, 0, P(40), 0, 0,0,0,0, 0, qA, 192, 1,0,1, 0);
    launch_conv(stream, Uc,  64, 0, 0, P(41), 0, 0,0,0,0, 0, qB, 192, 1,0,1, 0);
    win_attn_kernel<<<576, 128, 0, stream>>>(qB, qA, oA);   // attend(qB, kA, vA)
    win_attn_kernel<<<576, 128, 0, stream>>>(qA, qB, oB);   // attend(qA, kB, vB)
    launch_conv(stream, oA, 64, 0, 0, P(42), P(43), 0,0,0,0, CLS, u1, 64, 1,0,1, 0);
    launch_conv(stream, oB, 64, 0, 0, P(44), P(45), 0,0,0,0, Uc,  u2, 64, 1,0,1, 0);
    launch_conv(stream, u1, 64, u2, 64, P(52), P(53), 0,0,0,0, 0, UCo, 64, 3,1,1, 0);

    // ---- cross_attn ca2: xA = pansh, xB = U_pansharp -> UP ----
    launch_conv(stream, PSH, 64, 0, 0, P(46), 0, 0,0,0,0, 0, qA, 192, 1,0,1, 0);
    launch_conv(stream, Up,  64, 0, 0, P(47), 0, 0,0,0,0, 0, qB, 192, 1,0,1, 0);
    win_attn_kernel<<<576, 128, 0, stream>>>(qB, qA, oA);
    win_attn_kernel<<<576, 128, 0, stream>>>(qA, qB, oB);
    launch_conv(stream, oA, 64, 0, 0, P(48), P(49), 0,0,0,0, PSH, u1, 64, 1,0,1, 0);
    launch_conv(stream, oB, 64, 0, 0, P(50), P(51), 0,0,0,0, Up,  u2, 64, 1,0,1, 0);
    launch_conv(stream, u1, 64, u2, 64, P(54), P(55), 0,0,0,0, 0, UPo, 64, 3,1,1, 0);
#undef P
}
